// GatedMoEFFN_83468394431292
// MI455X (gfx1250) — matrix operand order repaired and checked by reading, not compiled, not run on hardware
//
#include <hip/hip_runtime.h>
#include <math.h>

// Problem constants (from reference): B=8, S=4096, D=512, F=2048, E=8
#define DD   512
#define FF   2048
#define EE   8
#define NTOK 32768   // B*S
#define MT   32      // tokens per workgroup tile (2 WMMA M-subtiles)

typedef __attribute__((ext_vector_type(16))) __bf16 v16bf;
typedef __attribute__((ext_vector_type(8)))  float  v8f;

union ABf16 { v16bf v; uint4 q[2]; };

__device__ __forceinline__ unsigned short f2bf(float f) {
  unsigned int u = __float_as_uint(f);
  u += 0x7FFFu + ((u >> 16) & 1u);   // round-to-nearest-even
  return (unsigned short)(u >> 16);
}

// -------- workspace layout (bytes) --------
static constexpr size_t XBF_OFF = 0;                                    // NTOK*D bf16 = 32 MB
static constexpr size_t W1T_OFF = XBF_OFF + (size_t)NTOK * DD * 2;      // [E,F,D] bf16 = 16 MB
static constexpr size_t W2T_OFF = W1T_OFF + (size_t)EE * FF * DD * 2;   // 16 MB
static constexpr size_t W3T_OFF = W2T_OFF + (size_t)EE * FF * DD * 2;   // [E,D,F] bf16 = 16 MB
static constexpr size_t WGT_OFF = W3T_OFF + (size_t)EE * DD * FF * 2;   // NTOK f32
static constexpr size_t CNT_OFF = WGT_OFF + (size_t)NTOK * 4;           // E ints (padded)
static constexpr size_t BKT_OFF = CNT_OFF + 256;                        // E*NTOK ints = 1 MB

// ------------------------------------------------------------------
__global__ void moe_init_kernel(int* counts, float* aux) {
  if (threadIdx.x < EE) counts[threadIdx.x] = 0;
  if (threadIdx.x == 0) aux[0] = 0.0f;   // aux_loss output
}

__global__ void moe_conv_x_kernel(const float* __restrict__ x,
                                  unsigned short* __restrict__ xb) {
  const int n = NTOK * DD;
  for (int i = blockIdx.x * blockDim.x + threadIdx.x; i < n;
       i += gridDim.x * blockDim.x)
    xb[i] = f2bf(x[i]);
}

// w1/w2: [E,D,F] fp32 -> [E,F,D] bf16 (K-contiguous per output column)
__global__ void moe_conv_wt_kernel(const float* __restrict__ w,
                                   unsigned short* __restrict__ wt) {
  const int n = EE * DD * FF;
  for (int i = blockIdx.x * blockDim.x + threadIdx.x; i < n;
       i += gridDim.x * blockDim.x) {
    int k  = i % DD;
    int nn = (i / DD) % FF;
    int e  = i / (DD * FF);
    wt[i] = f2bf(w[((size_t)e * DD + k) * FF + nn]);
  }
}

// w3: [E,F,D] fp32 -> [E,D,F] bf16
__global__ void moe_conv_w3t_kernel(const float* __restrict__ w,
                                    unsigned short* __restrict__ wt) {
  const int n = EE * DD * FF;
  for (int i = blockIdx.x * blockDim.x + threadIdx.x; i < n;
       i += gridDim.x * blockDim.x) {
    int f = i % FF;
    int d = (i / FF) % DD;
    int e = i / (DD * FF);
    wt[i] = f2bf(w[((size_t)e * FF + f) * DD + d]);
  }
}

// ------------------------------------------------------------------
// Gate: one wave (32 lanes) per token. 8 dot products of length 512,
// wave reduction, top-1 + softmax weight, atomic bucket append.
__global__ __launch_bounds__(256) void moe_gate_kernel(
    const float* __restrict__ x, const float* __restrict__ gw,
    const float* __restrict__ gb, float* __restrict__ wgt,
    int* __restrict__ counts, int* __restrict__ bucket) {
  int token = blockIdx.x * 8 + (threadIdx.x >> 5);
  int lane  = threadIdx.x & 31;
  if (token >= NTOK) return;

  const float* xr = x + (size_t)token * DD;
  float acc[EE];
#pragma unroll
  for (int e = 0; e < EE; ++e) acc[e] = 0.0f;

  for (int d = lane; d < DD; d += 32) {
    float xv = xr[d];
    const float* gwr = gw + (size_t)d * EE;
#pragma unroll
    for (int e = 0; e < EE; ++e) acc[e] += xv * gwr[e];
  }
#pragma unroll
  for (int e = 0; e < EE; ++e) {
#pragma unroll
    for (int off = 16; off > 0; off >>= 1)
      acc[e] += __shfl_xor(acc[e], off, 32);
    acc[e] += gb[e];
  }
  // top-1 (first max, matches argmax) + softmax value at argmax
  float mx = acc[0]; int top = 0;
#pragma unroll
  for (int e = 1; e < EE; ++e)
    if (acc[e] > mx) { mx = acc[e]; top = e; }
  float s = 0.0f;
#pragma unroll
  for (int e = 0; e < EE; ++e) s += expf(acc[e] - mx);

  if (lane == 0) {
    wgt[token] = 1.0f / s;                   // exp(mx-mx)/s
    int pos = atomicAdd(&counts[top], 1);
    bucket[top * NTOK + pos] = token;
  }
}

// ------------------------------------------------------------------
// Expert FFN: one workgroup per 32-token tile of one expert bucket.
// 8 waves: phase1 each wave makes 16 h-columns for both M-subtiles
// (2 GEMMs, K=512); phase2 each wave owns 64 output columns of
// y = h @ W3 (K=F) for both M-subtiles. Every B-operand register
// load feeds two WMMAs (M reuse) -> 2x arithmetic intensity vs M=16.
__global__ __launch_bounds__(256) void moe_ffn_kernel(
    const unsigned short* __restrict__ xb,
    const unsigned short* __restrict__ w1t,
    const unsigned short* __restrict__ w2t,
    const unsigned short* __restrict__ w3t,
    const float* __restrict__ b1, const float* __restrict__ b2,
    const float* __restrict__ b3, const float* __restrict__ wgt,
    const int* __restrict__ counts, const int* __restrict__ bucket,
    float* __restrict__ out) {
  __shared__ __align__(32) unsigned short x_lds[MT * DD];   // 32 KB
  __shared__ __align__(32) unsigned short h_lds[MT * 128];  // 8 KB
  __shared__ int   toks[MT];
  __shared__ float tw[MT];

  // map block -> (expert, tile)
  int e = 0, tile = 0, base = 0; bool found = false;
#pragma unroll
  for (int i = 0; i < EE; ++i) {
    int t = (counts[i] + MT - 1) >> 5;
    if (!found && (int)blockIdx.x < base + t) {
      e = i; tile = blockIdx.x - base; found = true;
    }
    base += t;
  }
  if (!found) return;   // uniform across the block
  int cnt = counts[e];

  int tid = threadIdx.x;
  if (tid < MT) {
    int idx = tile * MT + tid;
    int tok = (idx < cnt) ? bucket[e * NTOK + idx] : -1;
    toks[tid] = tok;
    tw[tid]   = (tok >= 0) ? wgt[tok] : 0.0f;
  }
  __syncthreads();

  // stage X tile [32, 512] bf16 (gathered rows); zeros for pad rows
  for (int i = tid; i < MT * DD / 8; i += 256) {
    int m = i >> 6;       // 64 uint4 chunks per row
    int c = i & 63;
    int tok = toks[m];
    uint4 v = make_uint4(0u, 0u, 0u, 0u);
    if (tok >= 0) v = *(const uint4*)(xb + (size_t)tok * DD + c * 8);
    *(uint4*)(x_lds + m * DD + c * 8) = v;
  }
  __syncthreads();

  const int wave  = tid >> 5;
  const int lane  = tid & 31;
  const int lrow  = lane & 15;                 // M (A) or N (B/C) index
  const int khalf = (lane < 16) ? 0 : 16;      // A-operand K split
  const int kb    = (lane < 16) ? 0 : 16;      // B-operand K split
  const int mhi   = (lane < 16) ? 0 : 8;       // C/D row half

  v8f yacc[2][4] = {};
  const unsigned short* w1b = w1t + (size_t)e * FF * DD;
  const unsigned short* w2b = w2t + (size_t)e * FF * DD;
  const unsigned short* w3b = w3t + (size_t)e * DD * FF;

  for (int chunk = 0; chunk < FF; chunk += 128) {
    // ---- phase 1: h[:, n0..n0+15] = gelu(X@W1+b1) * (X@W2+b2) ----
    int ncol = chunk + wave * 16 + lrow;
    const unsigned short* w1p = w1b + (size_t)ncol * DD;
    const unsigned short* w2p = w2b + (size_t)ncol * DD;
    v8f a1[2] = {}, a2[2] = {};
    for (int k0 = 0; k0 < DD; k0 += 32) {
      ABf16 A0, A1;
      A0.q[0] = *(const uint4*)(x_lds + lrow * DD + k0 + khalf);
      A0.q[1] = *(const uint4*)(x_lds + lrow * DD + k0 + khalf + 8);
      A1.q[0] = *(const uint4*)(x_lds + (lrow + 16) * DD + k0 + khalf);
      A1.q[1] = *(const uint4*)(x_lds + (lrow + 16) * DD + k0 + khalf + 8);
      v16bf B1 = *(const v16bf*)(w1p + k0 + kb);
      v16bf B2 = *(const v16bf*)(w2p + k0 + kb);
      a1[0] = __builtin_amdgcn_wmma_f32_16x16x32_bf16(false, A0.v, false, B1,
                                                      (short)0, a1[0], false, false);
      a1[1] = __builtin_amdgcn_wmma_f32_16x16x32_bf16(false, A1.v, false, B1,
                                                      (short)0, a1[1], false, false);
      a2[0] = __builtin_amdgcn_wmma_f32_16x16x32_bf16(false, A0.v, false, B2,
                                                      (short)0, a2[0], false, false);
      a2[1] = __builtin_amdgcn_wmma_f32_16x16x32_bf16(false, A1.v, false, B2,
                                                      (short)0, a2[1], false, false);
    }
    float bias1 = b1[e * FF + ncol];
    float bias2 = b2[e * FF + ncol];
#pragma unroll
    for (int mt = 0; mt < 2; ++mt) {
#pragma unroll
      for (int v = 0; v < 8; ++v) {
        float g  = a1[mt][v] + bias1;
        float hv = 0.5f * g * (1.0f + erff(g * 0.70710678f)) * (a2[mt][v] + bias2);
        int m = mt * 16 + v + mhi;
        h_lds[m * 128 + wave * 16 + lrow] = f2bf(hv);
      }
    }
    __syncthreads();

    // ---- phase 2: yacc += h_chunk @ W3[chunk rows, my 64 cols] ----
#pragma unroll
    for (int kk = 0; kk < 128; kk += 32) {
      ABf16 A0, A1;
      A0.q[0] = *(const uint4*)(h_lds + lrow * 128 + kk + khalf);
      A0.q[1] = *(const uint4*)(h_lds + lrow * 128 + kk + khalf + 8);
      A1.q[0] = *(const uint4*)(h_lds + (lrow + 16) * 128 + kk + khalf);
      A1.q[1] = *(const uint4*)(h_lds + (lrow + 16) * 128 + kk + khalf + 8);
      int kglob = chunk + kk + kb;
#pragma unroll
      for (int dt = 0; dt < 4; ++dt) {
        int d = wave * 64 + dt * 16 + lrow;
        v16bf Bv = *(const v16bf*)(w3b + (size_t)d * FF + kglob);
        yacc[0][dt] = __builtin_amdgcn_wmma_f32_16x16x32_bf16(
            false, A0.v, false, Bv, (short)0, yacc[0][dt], false, false);
        yacc[1][dt] = __builtin_amdgcn_wmma_f32_16x16x32_bf16(
            false, A1.v, false, Bv, (short)0, yacc[1][dt], false, false);
      }
    }
    __syncthreads();   // protect h_lds before next chunk overwrites it
  }

  // ---- epilogue: out[tok, d] = w_tok * (y + b3) ----
#pragma unroll
  for (int dt = 0; dt < 4; ++dt) {
    int d = wave * 64 + dt * 16 + lrow;
    float b3v = b3[e * DD + d];
#pragma unroll
    for (int mt = 0; mt < 2; ++mt) {
#pragma unroll
      for (int v = 0; v < 8; ++v) {
        int m = mt * 16 + v + mhi;
        int tok = toks[m];
        if (tok >= 0)
          out[(size_t)tok * DD + d] = tw[m] * (yacc[mt][dt][v] + b3v);
      }
    }
  }
}

// ------------------------------------------------------------------
extern "C" void kernel_launch(void* const* d_in, const int* in_sizes, int n_in,
                              void* d_out, int out_size, void* d_ws, size_t ws_size,
                              hipStream_t stream) {
  const float* x  = (const float*)d_in[0];
  const float* gw = (const float*)d_in[1];
  const float* gb = (const float*)d_in[2];
  const float* w1 = (const float*)d_in[3];
  const float* b1 = (const float*)d_in[4];
  const float* w2 = (const float*)d_in[5];
  const float* b2 = (const float*)d_in[6];
  const float* w3 = (const float*)d_in[7];
  const float* b3 = (const float*)d_in[8];
  float* out = (float*)d_out;

  char* ws = (char*)d_ws;
  unsigned short* xbuf = (unsigned short*)(ws + XBF_OFF);
  unsigned short* w1t  = (unsigned short*)(ws + W1T_OFF);
  unsigned short* w2t  = (unsigned short*)(ws + W2T_OFF);
  unsigned short* w3t  = (unsigned short*)(ws + W3T_OFF);
  float* wgt  = (float*)(ws + WGT_OFF);
  int* counts = (int*)(ws + CNT_OFF);
  int* bucket = (int*)(ws + BKT_OFF);

  moe_init_kernel<<<1, 32, 0, stream>>>(counts, out + (size_t)NTOK * DD);
  moe_conv_x_kernel<<<2048, 256, 0, stream>>>(x, xbuf);
  moe_conv_wt_kernel<<<2048, 256, 0, stream>>>(w1, w1t);
  moe_conv_wt_kernel<<<2048, 256, 0, stream>>>(w2, w2t);
  moe_conv_w3t_kernel<<<2048, 256, 0, stream>>>(w3, w3t);
  moe_gate_kernel<<<NTOK / 8, 256, 0, stream>>>(x, gw, gb, wgt, counts, bucket);
  moe_ffn_kernel<<<NTOK / MT + EE, 256, 0, stream>>>(
      xbuf, w1t, w2t, w3t, b1, b2, b3, wgt, counts, bucket, out);
}
